// CausalGraphAttention_89790586290220
// MI455X (gfx1250) — matrix operand order repaired and checked by reading, not compiled, not run on hardware
//
#include <hip/hip_runtime.h>
#include <hip/hip_bf16.h>
#include <math.h>

// ---------------------------------------------------------------------------
// Causal multi-head attention, CDNA5 (gfx1250) wave32 WMMA pipeline.
// Round 5: sched_barrier keeps bias epilogue loads out of the GEMM k-loop
// (kills the folded scratch spills seen in round 4).
//  * attention: 4 waves/block share one head + 4 consecutive query tiles;
//    K/V chunks staged once per block into double-buffered LDS via the
//    gfx1250 async global->LDS path (ASYNCcnt), 4x less L2 traffic.
//  * GEMMs: 32x64 tile per wave, software-pipelined fragment loads.
// ---------------------------------------------------------------------------

typedef __attribute__((ext_vector_type(16))) _Float16 v16h;
typedef __attribute__((ext_vector_type(8)))  _Float16 v8h;
typedef __attribute__((ext_vector_type(8)))  float    v8f;
typedef __attribute__((ext_vector_type(4)))  int      v4i;

#define NTOK 4096
#define CH   512
#define NH   8
#define HD   64

#define HAS_ASYNC_LDS __has_builtin(__builtin_amdgcn_global_load_async_to_lds_b128)

__device__ __forceinline__ void sched_fence() {
#if __has_builtin(__builtin_amdgcn_sched_barrier)
  __builtin_amdgcn_sched_barrier(0);
#endif
}

// ---- fragment helpers (layouts per CDNA5 ISA 7.12.2, wave32) --------------
__device__ __forceinline__ v16h cat16(v8h lo, v8h hi) {
  return __builtin_shufflevector(lo, hi, 0,1,2,3,4,5,6,7,8,9,10,11,12,13,14,15);
}
// A-matrix 16x32 f16: lane holds row M=lane%16; elems j<8 -> K=half*8+j,
// j>=8 -> K=16+half*8+(j-8).
__device__ __forceinline__ v16h load_afrag(const _Float16* base, int half) {
  v8h lo = *(const v8h*)(base + half * 8);
  v8h hi = *(const v8h*)(base + 16 + half * 8);
  return cat16(lo, hi);
}
// B-matrix 32x16 f16: lane holds col N=lane%16; elems j<8 -> K=half*8+j,
// j>=8 -> K=16+half*8+(j-8).
__device__ __forceinline__ v16h load_bfrag(const _Float16* base, int half) {
  v8h lo = *(const v8h*)(base + half * 8);
  v8h hi = *(const v8h*)(base + 16 + half * 8);
  return cat16(lo, hi);
}
__device__ __forceinline__ v8f wmma32(v16h a, v16h b, v8f c) {
  return __builtin_amdgcn_wmma_f32_16x16x32_f16(false, a, false, b, (short)0, c,
                                                false, false);
}

// ---- 16B copy: global -> LDS (async DMA if available) ---------------------
#if HAS_ASYNC_LDS
typedef __attribute__((address_space(1))) v4i* as1_v4i_p;
typedef __attribute__((address_space(3))) v4i* as3_v4i_p;
#endif
__device__ __forceinline__ void cp16_g2l(const _Float16* g, _Float16* l) {
#if HAS_ASYNC_LDS
  __builtin_amdgcn_global_load_async_to_lds_b128((as1_v4i_p)g, (as3_v4i_p)l,
                                                 0, 0);
#else
  *(uint4*)l = *(const uint4*)g;
#endif
}
__device__ __forceinline__ void wait_async4() {
#if HAS_ASYNC_LDS
#if __has_builtin(__builtin_amdgcn_s_wait_asynccnt)
  __builtin_amdgcn_s_wait_asynccnt(4);
#else
  asm volatile("s_wait_asynccnt 0x4" ::: "memory");
#endif
#endif
}
__device__ __forceinline__ void wait_async0() {
#if HAS_ASYNC_LDS
#if __has_builtin(__builtin_amdgcn_s_wait_asynccnt)
  __builtin_amdgcn_s_wait_asynccnt(0);
#else
  asm volatile("s_wait_asynccnt 0x0" ::: "memory");
#endif
#endif
}

// ---- f32 -> f16 convert ----------------------------------------------------
__global__ void cvt_f16_kernel(const float* __restrict__ in,
                               _Float16* __restrict__ out, int n) {
  int i = blockIdx.x * blockDim.x + threadIdx.x;
  if (i < n) out[i] = (_Float16)in[i];
}

// ---- GEMM core: C[32x64] = A[32x512] * W[64rows x 512]^T + bias -----------
// A row-major [M, 512] f16, W row-major [N, 512] f16 (A*W^T).
// c[rb*4+t] = rows rb*16.., cols t*16..
__device__ __forceinline__ void gemm_tile_32x64(const _Float16* __restrict__ A,
                                                const _Float16* __restrict__ W,
                                                const float* __restrict__ bias,
                                                int m0, int n0, v8f c[8]) {
  const int lane = threadIdx.x & 31;
  const int half = lane >> 4;
  const int lm   = lane & 15;
#pragma unroll
  for (int i = 0; i < 8; ++i) c[i] = {};
  const _Float16* ar0 = A + (size_t)(m0 + lm) * CH;
  const _Float16* ar1 = ar0 + (size_t)16 * CH;
  const _Float16* w0 = W + (size_t)(n0 + 0 * 16 + lm) * CH;
  const _Float16* w1 = W + (size_t)(n0 + 1 * 16 + lm) * CH;
  const _Float16* w2 = W + (size_t)(n0 + 2 * 16 + lm) * CH;
  const _Float16* w3 = W + (size_t)(n0 + 3 * 16 + lm) * CH;

  v16h a0 = load_afrag(ar0, half);
  v16h a1 = load_afrag(ar1, half);
  v16h b0 = load_bfrag(w0, half);
  v16h b1 = load_bfrag(w1, half);
  v16h b2 = load_bfrag(w2, half);
  v16h b3 = load_bfrag(w3, half);

#pragma unroll
  for (int kc = 0; kc < CH / 32; ++kc) {
    v16h a0n, a1n, bn0, bn1, bn2, bn3;
    if (kc < CH / 32 - 1) {
      const int kb = (kc + 1) * 32;   // issue next chunk's loads first
      a0n = load_afrag(ar0 + kb, half);
      a1n = load_afrag(ar1 + kb, half);
      bn0 = load_bfrag(w0 + kb, half);
      bn1 = load_bfrag(w1 + kb, half);
      bn2 = load_bfrag(w2 + kb, half);
      bn3 = load_bfrag(w3 + kb, half);
    } else {
      a0n = a0; a1n = a1; bn0 = b0; bn1 = b1; bn2 = b2; bn3 = b3;
    }
    c[0] = wmma32(a0, b0, c[0]);
    c[1] = wmma32(a0, b1, c[1]);
    c[2] = wmma32(a0, b2, c[2]);
    c[3] = wmma32(a0, b3, c[3]);
    c[4] = wmma32(a1, b0, c[4]);
    c[5] = wmma32(a1, b1, c[5]);
    c[6] = wmma32(a1, b2, c[6]);
    c[7] = wmma32(a1, b3, c[7]);
    a0 = a0n; a1 = a1n; b0 = bn0; b1 = bn1; b2 = bn2; b3 = bn3;
  }
  sched_fence();  // keep epilogue bias loads out of the pipelined k-loop
#pragma unroll
  for (int t = 0; t < 4; ++t) {
    float bv = bias[n0 + t * 16 + lm];
#pragma unroll
    for (int r = 0; r < 8; ++r) { c[t][r] += bv; c[4 + t][r] += bv; }
  }
}

// ---- QKV projection: q16/k16 row-major [N,512] f16, V transposed ----------
__global__ void qkv_gemm_kernel(const _Float16* __restrict__ x16,
                                const _Float16* __restrict__ wq,
                                const _Float16* __restrict__ wk,
                                const _Float16* __restrict__ wv,
                                const float* __restrict__ bq,
                                const float* __restrict__ bk,
                                const float* __restrict__ bv,
                                _Float16* __restrict__ q16,
                                _Float16* __restrict__ k16,
                                _Float16* __restrict__ vt16) {
  const int wid = blockIdx.x * (blockDim.x >> 5) + (threadIdx.x >> 5);
  const int tn  = wid & 7;           // 8 tiles of 64 cols
  const int tm  = (wid >> 3) & 127;  // 128 tiles of 32 rows
  const int mat = wid >> 10;         // 0=Q 1=K 2=V
  const _Float16* W = (mat == 0) ? wq : (mat == 1) ? wk : wv;
  const float*    b = (mat == 0) ? bq : (mat == 1) ? bk : bv;

  v8f c[8];
  gemm_tile_32x64(x16, W, b, tm * 32, tn * 64, c);

  const int lane = threadIdx.x & 31;
  const int half = lane >> 4;
  const int lm   = lane & 15;
  if (mat < 2) {
    _Float16* out = (mat == 0) ? q16 : k16;
#pragma unroll
    for (int rb = 0; rb < 2; ++rb)
#pragma unroll
      for (int t = 0; t < 4; ++t)
#pragma unroll
        for (int r = 0; r < 8; ++r)
          out[(size_t)(tm * 32 + rb * 16 + half * 8 + r) * CH + tn * 64 +
              t * 16 + lm] = (_Float16)c[rb * 4 + t][r];
  } else {
    // transposed store: vt[d][n], rows (r) contiguous -> b128 stores
#pragma unroll
    for (int rb = 0; rb < 2; ++rb)
#pragma unroll
      for (int t = 0; t < 4; ++t) {
        v8h pk;
#pragma unroll
        for (int r = 0; r < 8; ++r) pk[r] = (_Float16)c[rb * 4 + t][r];
        *(v8h*)(vt16 + (size_t)(tn * 64 + t * 16 + lm) * NTOK + tm * 32 +
                rb * 16 + half * 8) = pk;
      }
  }
}

// ---- causal flash attention ------------------------------------------------
// Block = 4 waves = one head x 4 consecutive 16-query tiles. K/V 32-key
// chunks staged once per block into double-buffered LDS (async DMA).
__global__ void attn_kernel(const _Float16* __restrict__ q16,
                            const _Float16* __restrict__ k16,
                            const _Float16* __restrict__ vt16,
                            _Float16* __restrict__ ao16) {
  __shared__ alignas(16) _Float16 Kt[2][32 * 64];  // [key][d]   4KB each
  __shared__ alignas(16) _Float16 Vt[2][64 * 32];  // [d][key]   4KB each
  __shared__ alignas(16) _Float16 Pt[4][16 * 32];  // per-wave P transpose

  const int tid   = threadIdx.x;
  const int wslot = tid >> 5;
  const int h     = blockIdx.x & (NH - 1);
  const int qg    = blockIdx.x >> 3;        // 0..63 query groups of 64
  const int qt    = qg * 4 + wslot;         // this wave's 16-query tile
  const int lane  = tid & 31;
  const int half  = lane >> 4;
  const int lm    = lane & 15;
  _Float16* pbuf  = &Pt[wslot][0];

  // Q A-fragments (16 rows x 64 dims), hoisted
  const _Float16* qp = q16 + (size_t)(qt * 16 + lm) * CH + h * HD;
  v16h aQ0 = load_afrag(qp, half);
  v16h aQ1 = load_afrag(qp + 32, half);

  v8f O[4]; O[0] = {}; O[1] = {}; O[2] = {}; O[3] = {};
  float rmax[8], rsum[8];
#pragma unroll
  for (int r = 0; r < 8; ++r) { rmax[r] = -__builtin_inff(); rsum[r] = 0.f; }

  const int qbase   = qt * 16 + half * 8;     // query row = qbase + r
  const int my_nkc  = (qt * 16 + 15) / 32 + 1;
  const int nkc_blk = 2 * qg + 2;             // chunks for the whole block

  // --- cooperative stage of one 32-key K/V chunk (4 x 16B per thread) ----
  auto stage = [&](int buf, int kc) {
    const int kb = kc * 32;
#pragma unroll
    for (int i = 0; i < 2; ++i) {
      int s = tid * 2 + i;                 // 256 16B segments each
      {                                    // K tile: 32 rows x 128B
        int row = s >> 3, off = (s & 7) * 8;
        cp16_g2l(k16 + (size_t)(kb + row) * CH + h * HD + off,
                 &Kt[buf][row * 64 + off]);
      }
      {                                    // V tile: 64 rows x 64B
        int row = s >> 2, off = (s & 3) * 8;
        cp16_g2l(vt16 + (size_t)(h * HD + row) * NTOK + kb + off,
                 &Vt[buf][row * 32 + off]);
      }
    }
  };

  stage(0, 0);
  for (int kc = 0; kc < nkc_blk; ++kc) {
    if (kc + 1 < nkc_blk) { stage((kc + 1) & 1, kc + 1); wait_async4(); }
    else                  { wait_async0(); }
    __syncthreads();                       // chunk kc visible to all waves

    if (kc < my_nkc) {                     // wave-uniform guard, EXEC all-1s
      const int buf = kc & 1;
      const int kb  = kc * 32;
      // ---- S = Q K^T (16 x 32) from LDS ---------------------------------
      v8f S[2]; S[0] = {}; S[1] = {};
#pragma unroll
      for (int g = 0; g < 2; ++g) {
        const _Float16* kp = &Kt[buf][(g * 16 + lm) * 64];
        S[g] = wmma32(aQ0, load_bfrag(kp, half), S[g]);
        S[g] = wmma32(aQ1, load_bfrag(kp + 32, half), S[g]);
      }
#pragma unroll
      for (int g = 0; g < 2; ++g) {
        const int key = kb + g * 16 + lm;
#pragma unroll
        for (int r = 0; r < 8; ++r) {
          float s = S[g][r] * 0.125f;      // 1/sqrt(64)
          S[g][r] = (key <= qbase + r) ? s : -__builtin_inff();
        }
      }
      // ---- online softmax (16-lane xor reductions) ----------------------
      float cm[8];
#pragma unroll
      for (int r = 0; r < 8; ++r) cm[r] = fmaxf(S[0][r], S[1][r]);
#pragma unroll
      for (int off = 1; off < 16; off <<= 1)
#pragma unroll
        for (int r = 0; r < 8; ++r)
          cm[r] = fmaxf(cm[r], __shfl_xor(cm[r], off, 32));
      float alpha[8];
#pragma unroll
      for (int r = 0; r < 8; ++r) {
        float nm = fmaxf(rmax[r], cm[r]);
        alpha[r] = __expf(rmax[r] - nm);
        rmax[r]  = nm;
      }
      float rs[8];
#pragma unroll
      for (int r = 0; r < 8; ++r) {
        float p0 = __expf(S[0][r] - rmax[r]);
        float p1 = __expf(S[1][r] - rmax[r]);
        S[0][r] = p0; S[1][r] = p1;
        rs[r] = p0 + p1;
      }
#pragma unroll
      for (int off = 1; off < 16; off <<= 1)
#pragma unroll
        for (int r = 0; r < 8; ++r) rs[r] += __shfl_xor(rs[r], off, 32);
#pragma unroll
      for (int r = 0; r < 8; ++r) rsum[r] = rsum[r] * alpha[r] + rs[r];
#pragma unroll
      for (int t = 0; t < 4; ++t)
#pragma unroll
        for (int r = 0; r < 8; ++r) O[t][r] *= alpha[r];

      // ---- transpose P (C-layout -> A-layout) through per-wave LDS ------
#pragma unroll
      for (int g = 0; g < 2; ++g)
#pragma unroll
        for (int r = 0; r < 8; ++r)
          pbuf[(half * 8 + r) * 32 + g * 16 + lm] = (_Float16)S[g][r];
      asm volatile("" ::: "memory");       // same-wave LDS is in-order
      v16h aP = load_afrag(pbuf + lm * 32, half);

      // ---- O += P @ V (V from LDS) --------------------------------------
#pragma unroll
      for (int t = 0; t < 4; ++t)
        O[t] = wmma32(aP, load_bfrag(&Vt[buf][(t * 16 + lm) * 32], half),
                      O[t]);
    }
    __syncthreads();                       // done reading buf before restage
  }

  // ---- normalize and store attention output (row-major f16) -------------
#pragma unroll
  for (int t = 0; t < 4; ++t)
#pragma unroll
    for (int r = 0; r < 8; ++r) {
      float v = O[t][r] / rsum[r];
      ao16[(size_t)(qt * 16 + half * 8 + r) * CH + h * HD + t * 16 + lm] =
          (_Float16)v;
    }
}

// ---- output projection: f32 out = ao16 @ Wo^T + bo ------------------------
__global__ void out_gemm_kernel(const _Float16* __restrict__ ao16,
                                const _Float16* __restrict__ wo,
                                const float* __restrict__ bo,
                                float* __restrict__ out) {
  const int wid = blockIdx.x * (blockDim.x >> 5) + (threadIdx.x >> 5);
  const int tn  = wid & 7;
  const int tm  = (wid >> 3) & 127;
  v8f c[8];
  gemm_tile_32x64(ao16, wo, bo, tm * 32, tn * 64, c);
  const int lane = threadIdx.x & 31;
  const int half = lane >> 4;
  const int lm   = lane & 15;
#pragma unroll
  for (int rb = 0; rb < 2; ++rb)
#pragma unroll
    for (int t = 0; t < 4; ++t)
#pragma unroll
      for (int r = 0; r < 8; ++r)
        out[(size_t)(tm * 32 + rb * 16 + half * 8 + r) * CH + tn * 64 +
            t * 16 + lm] = c[rb * 4 + t][r];
}

// ---------------------------------------------------------------------------
extern "C" void kernel_launch(void* const* d_in, const int* in_sizes, int n_in,
                              void* d_out, int out_size, void* d_ws,
                              size_t ws_size, hipStream_t stream) {
  const float* x  = (const float*)d_in[0];
  // d_in[1] edge_index (unused by reference), d_in[2] temporal_mask (causal)
  const float* Wq = (const float*)d_in[3];
  const float* bq = (const float*)d_in[4];
  const float* Wk = (const float*)d_in[5];
  const float* bk = (const float*)d_in[6];
  const float* Wv = (const float*)d_in[7];
  const float* bv = (const float*)d_in[8];
  const float* Wo = (const float*)d_in[9];
  const float* bo = (const float*)d_in[10];
  float* out = (float*)d_out;

  // workspace layout (bytes)
  char* ws = (char*)d_ws;
  const size_t SZ_X = (size_t)NTOK * CH * sizeof(_Float16); // 4 MB
  const size_t SZ_W = (size_t)CH * CH * sizeof(_Float16);   // 512 KB
  _Float16* x16  = (_Float16*)(ws);
  _Float16* wq16 = (_Float16*)(ws + SZ_X);
  _Float16* wk16 = (_Float16*)(ws + SZ_X + SZ_W);
  _Float16* wv16 = (_Float16*)(ws + SZ_X + 2 * SZ_W);
  _Float16* wo16 = (_Float16*)(ws + SZ_X + 3 * SZ_W);
  _Float16* q16  = (_Float16*)(ws + SZ_X + 4 * SZ_W);
  _Float16* k16  = (_Float16*)(ws + 2 * SZ_X + 4 * SZ_W);
  _Float16* vt16 = (_Float16*)(ws + 3 * SZ_X + 4 * SZ_W);
  _Float16* ao16 = (_Float16*)(ws + 4 * SZ_X + 4 * SZ_W); // total ~22 MB

  const int NXE = NTOK * CH;  // 2097152
  const int NWE = CH * CH;    // 262144
  cvt_f16_kernel<<<NXE / 256, 256, 0, stream>>>(x, x16, NXE);
  cvt_f16_kernel<<<NWE / 256, 256, 0, stream>>>(Wq, wq16, NWE);
  cvt_f16_kernel<<<NWE / 256, 256, 0, stream>>>(Wk, wk16, NWE);
  cvt_f16_kernel<<<NWE / 256, 256, 0, stream>>>(Wv, wv16, NWE);
  cvt_f16_kernel<<<NWE / 256, 256, 0, stream>>>(Wo, wo16, NWE);

  // QKV: 128 m-tiles * 8 n-tiles * 3 matrices = 3072 waves, 4 waves/block
  qkv_gemm_kernel<<<768, 128, 0, stream>>>(x16, wq16, wk16, wv16, bq, bk, bv,
                                           q16, k16, vt16);
  // attention: 8 heads * 64 query groups = 512 blocks (4 waves each)
  attn_kernel<<<512, 128, 0, stream>>>(q16, k16, vt16, ao16);
  // output projection: 1024 waves
  out_gemm_kernel<<<256, 128, 0, stream>>>(ao16, wo16, bo, out);
}
